// AgentNet_59115929862544
// MI455X (gfx1250) — matrix operand order repaired and checked by reading, not compiled, not run on hardware
//
#include <hip/hip_runtime.h>
#include <hip/hip_bf16.h>

typedef __attribute__((ext_vector_type(16))) _Float16 v16h;
typedef __attribute__((ext_vector_type(8)))  float    v8f;

#define M_ROWS 1000000

// CDNA5 transcendental tanh. TRANS ops need 1 independent op before the
// result is consumed (ISA 7.4); inline asm bypasses compiler hazard
// tracking, so pad with v_nop inside the asm.
__device__ __forceinline__ float fast_tanh(float x) {
  float r;
  asm volatile("v_tanh_f32 %0, %1\n\tv_nop" : "=v"(r) : "v"(x));
  return r;
}

// ---------------- Pass 1: sum(x) over 1M rows (x = state0[1:]) -------------
__global__ __launch_bounds__(256) void k_reduce(const float* __restrict__ state0,
                                                float* __restrict__ partials) {
  __shared__ float r0[256], r1[256];
  const int t = threadIdx.x;
  const int gid = blockIdx.x * 256 + t;
  const float2* x = (const float2*)state0;   // row i of x is state0 row i+1
  float s0 = 0.f, s1 = 0.f;
  for (int i = gid; i < M_ROWS; i += 256 * 256) {
    float2 v = x[i + 1];
    s0 += v.x; s1 += v.y;
  }
  r0[t] = s0; r1[t] = s1;
  __syncthreads();
  for (int off = 128; off > 0; off >>= 1) {
    if (t < off) { r0[t] += r0[t + off]; r1[t] += r1[t + off]; }
    __syncthreads();
  }
  if (t == 0) {
    partials[2 * blockIdx.x + 0] = r0[0];
    partials[2 * blockIdx.x + 1] = r1[0];
  }
}

// ---------------- Pass 2: h, c = Wh*h + W_b, and f16 hi/lo B-fragments -----
__global__ __launch_bounds__(128) void k_prep(const float* __restrict__ partials,
                                              const float* __restrict__ pt_sc,
                                              const float* __restrict__ embed_w,
                                              const float* __restrict__ embed_b,
                                              const float* __restrict__ W_w,
                                              const float* __restrict__ W_b,
                                              float* __restrict__ c_vec,
                                              _Float16* __restrict__ Bfrag) {
  __shared__ float r0[128], r1[128];
  __shared__ float h[34];
  __shared__ float mx[2];
  const int t = threadIdx.x;
  r0[t] = partials[2 * t]     + partials[2 * (t + 128)];
  r1[t] = partials[2 * t + 1] + partials[2 * (t + 128) + 1];
  __syncthreads();
  for (int off = 64; off > 0; off >>= 1) {
    if (t < off) { r0[t] += r0[t + off]; r1[t] += r1[t + off]; }
    __syncthreads();
  }
  if (t == 0) { mx[0] = r0[0] * (1.0f / M_ROWS); mx[1] = r1[0] * (1.0f / M_ROWS); }
  __syncthreads();
  // e.mean = embed_w @ mean(x) + embed_b  (affinity of e in x)
  if (t < 32) h[t] = embed_w[2 * t] * mx[0] + embed_w[2 * t + 1] * mx[1] + embed_b[t];
  if (t < 2)  h[32 + t] = pt_sc[t];
  __syncthreads();
  // c[j] = W_b[j] + Wh[j,:] . h   (Wh = W_w[:,32:66], row-major stride 66)
  float acc = W_b[t];
  for (int k = 0; k < 34; ++k) acc += W_w[t * 66 + 32 + k] * h[k];
  c_vec[t] = acc;
  // Pre-swizzle We (= W_w[:, :32]) into dense f16 B fragments.
  // B layout (32x16, wave32): lane&15 = N, half index hh -> K = 8*(lane>>4) + hh + (hh>=8 ? 8 : 0). Chunk c covers
  // output neurons j = 16c..16c+15. hi part at [0,4096), lo at [4096,8192).
  for (int idx = t; idx < 4096; idx += 128) {
    const int chunk = idx >> 9;
    const int lane  = (idx >> 4) & 31;
    const int hh    = idx & 15;
    const int K     = ((lane >> 4) << 3) + hh + ((hh & 8) ? 8 : 0);
    const float w   = W_w[(16 * chunk + (lane & 15)) * 66 + K];
    const _Float16 whi = (_Float16)w;
    Bfrag[idx]        = whi;
    Bfrag[4096 + idx] = (_Float16)(w - (float)whi);
  }
}

// ---------------- Pass 3: per-row MLP via WMMA -----------------------------
__global__ __launch_bounds__(256) void k_main(const float* __restrict__ state0,
                                              const float* __restrict__ V_w,
                                              const float* __restrict__ V_b,
                                              const float* __restrict__ c_vec,
                                              const _Float16* __restrict__ Bfrag,
                                              const float* __restrict__ embed_w,
                                              const float* __restrict__ embed_b,
                                              float* __restrict__ out) {
  __shared__ float s_embed[96];   // embed_w (64) then embed_b (32)
  __shared__ float s_c[128];
  __shared__ float s_v[128];
  const int tid = threadIdx.x;
  if (tid < 96)  s_embed[tid] = (tid < 64) ? embed_w[tid] : embed_b[tid - 64];
  if (tid < 128) { s_c[tid] = c_vec[tid]; s_v[tid] = V_w[tid]; }
  __syncthreads();

  const int lane = tid & 31;
  const int tile = blockIdx.x * 8 + (tid >> 5);   // 16 rows per wave
  const int rowBase = tile * 16;
  if (rowBase >= M_ROWS) return;                  // wave-uniform

  const float2* x = (const float2*)state0;
  const int mrow = lane & 15;
  const float2 xv = x[rowBase + mrow + 1];
  __builtin_prefetch(&x[rowBase + mrow + 1 + 2048], 0, 1);  // stream ahead

  // Build A fragments (16-bit A 16x32 layout): lane -> M = lane&15,
  // half hh -> K = kbase + hh + (hh>=8 ? 8 : 0), kbase = (lane<16) ? 0 : 8.
  const int kbase = (lane < 16) ? 0 : 8;
  v16h a_hi, a_lo;
#pragma unroll
  for (int hh = 0; hh < 16; ++hh) {
    const int K = kbase + hh + ((hh & 8) ? 8 : 0);
    const float ev = xv.x * s_embed[2 * K] + xv.y * s_embed[2 * K + 1] + s_embed[64 + K];
    const _Float16 eh = (_Float16)ev;
    a_hi[hh] = eh;
    a_lo[hh] = (_Float16)(ev - (float)eh);
  }

  const int n = lane & 15;                  // D-matrix column for this lane
  const v16h* Bh = (const v16h*)Bfrag;      // [chunk*32 + lane]
  const v16h* Bl = (const v16h*)(Bfrag + 4096);
  float partial[8] = {0.f, 0.f, 0.f, 0.f, 0.f, 0.f, 0.f, 0.f};

#pragma unroll
  for (int c = 0; c < 8; ++c) {
    const v16h b_hi = Bh[c * 32 + lane];
    const v16h b_lo = Bl[c * 32 + lane];
    v8f acc = {};
    // fp32-accurate split product: hi*hi + lo*hi + hi*lo (lo*lo ~ 2^-22, dropped)
    acc = __builtin_amdgcn_wmma_f32_16x16x32_f16(false, a_hi, false, b_hi, (short)0, acc, false, false);
    acc = __builtin_amdgcn_wmma_f32_16x16x32_f16(false, a_lo, false, b_hi, (short)0, acc, false, false);
    acc = __builtin_amdgcn_wmma_f32_16x16x32_f16(false, a_hi, false, b_lo, (short)0, acc, false, false);
    const float cj = s_c[c * 16 + n];
    const float vj = s_v[c * 16 + n];
#pragma unroll
    for (int v = 0; v < 8; ++v) {           // D: M = v + 8*(lane>=16), N = n
      const float z = fast_tanh(acc[v] + cj);
      partial[v] += z * vj;
    }
  }

  // Reduce over the 16 N-lanes within each half-wave (butterfly).
#pragma unroll
  for (int m = 8; m > 0; m >>= 1) {
#pragma unroll
    for (int v = 0; v < 8; ++v) partial[v] += __shfl_xor(partial[v], m, 32);
  }

  if (n == 0) {                              // lanes 0 and 16 write 8 rows each
    const float vb = V_b[0];
    const int base = rowBase + ((lane >> 4) << 3);
    float res[8];
#pragma unroll
    for (int v = 0; v < 8; ++v)
      res[v] = 1.0f / (1.0f + __expf(-(partial[v] + vb)));
    float4* o4 = (float4*)(out + base);
    o4[0] = make_float4(res[0], res[1], res[2], res[3]);
    o4[1] = make_float4(res[4], res[5], res[6], res[7]);
  }
}

extern "C" void kernel_launch(void* const* d_in, const int* in_sizes, int n_in,
                              void* d_out, int out_size, void* d_ws, size_t ws_size,
                              hipStream_t stream) {
  const float* state0  = (const float*)d_in[0];   // (M+1, 2)
  const float* pt_sc   = (const float*)d_in[1];   // (2,)
  const float* embed_w = (const float*)d_in[2];   // (32, 2)
  const float* embed_b = (const float*)d_in[3];   // (32,)
  const float* W_w     = (const float*)d_in[4];   // (128, 66)
  const float* W_b     = (const float*)d_in[5];   // (128,)
  const float* V_w     = (const float*)d_in[6];   // (1, 128)
  const float* V_b     = (const float*)d_in[7];   // (1,)
  float* out = (float*)d_out;                     // (M, 1) fp32

  char* ws = (char*)d_ws;
  float*    c_vec    = (float*)ws;                  // 128 f32   =   512 B
  _Float16* Bfrag    = (_Float16*)(ws + 512);       // 8192 f16  = 16384 B
  float*    partials = (float*)(ws + 512 + 16384);  // 512 f32   =  2048 B

  k_reduce<<<256, 256, 0, stream>>>(state0, partials);
  k_prep<<<1, 128, 0, stream>>>(partials, pt_sc, embed_w, embed_b, W_w, W_b,
                                c_vec, Bfrag);
  const int tiles  = (M_ROWS + 15) / 16;   // 62500 waves
  const int blocks = (tiles + 7) / 8;      // 8 waves / block
  k_main<<<blocks, 256, 0, stream>>>(state0, V_w, V_b, c_vec, Bfrag,
                                     embed_w, embed_b, out);
}
